// DRQ_2448131359005
// MI455X (gfx1250) — matrix operand order repaired and checked by reading, not compiled, not run on hardware
//
#include <hip/hip_runtime.h>
#include <hip/hip_bf16.h>
#include <stdint.h>

// DRQ (residual vector quantization, M=4 stages) for MI455X / gfx1250.
// Compute-bound (~137 GFLOP vs ~18MB HBM traffic) -> bf16 WMMA for both
// matmuls, full codebook resident in LDS (row-major + transposed copies).

#define NROWS  131072
#define DDIM   32
#define KCB    2048
#define MSTAGE 4

typedef __bf16 v16bf __attribute__((ext_vector_type(16)));
typedef float  v8f   __attribute__((ext_vector_type(8)));

union BF16x16 {
  v16bf v;
  unsigned short s[16];
  uint4 q[2];
};

__device__ __forceinline__ unsigned short f2bf(float f) {
  union { float f; unsigned u; } c; c.f = f;
  unsigned u = c.u;
  return (unsigned short)((u + 0x7fffu + ((u >> 16) & 1u)) >> 16);  // RNE
}
__device__ __forceinline__ float bf2f(unsigned short b) {
  union { unsigned u; float f; } c; c.u = ((unsigned)b) << 16;
  return c.f;
}

__global__ void __launch_bounds__(256)
drq_kernel(const float* __restrict__ x, const float* __restrict__ cbG,
           const float* __restrict__ scale, float* __restrict__ out) {
  // ---- LDS: 322048 B total (<= 320KB/WGP) ----
  __shared__ __align__(16) unsigned short cbB[KCB * DDIM];   // row-major bf16
  __shared__ __align__(16) unsigned short cbT[DDIM * KCB];   // transposed bf16
  __shared__ float cn2[KCB];                                  // ||c||^2 (f32)
  __shared__ float res[8][16][32];                            // residual per wave
  __shared__ float dlt[8][16][32];                            // qhard - qsoft
  __shared__ __align__(16) float pS[8][16][32];               // prob / soft scratch
  __shared__ int   codesS[8][MSTAGE][16];
  __shared__ float lS[8][16];

  const int t    = threadIdx.x;
  const int w    = t >> 5;        // wave id (8 waves, 16 rows each)
  const int lane = t & 31;
  const int lp   = lane & 15;     // lane within half
  const int h    = lane >> 4;     // half id

  // ---- cooperative init: bf16 codebook (both layouts) + ||c||^2 ----
  for (int k = t; k < KCB; k += 256) {
    const float* c = cbG + k * DDIM;
    float ss = 0.f;
    for (int d = 0; d < DDIM; ++d) {
      float v = c[d];
      ss += v * v;
      unsigned short b = f2bf(v);
      cbB[k * DDIM + d] = b;
      cbT[d * KCB + k]  = b;
    }
    cn2[k] = ss;
  }
  // residual = x tile; delta = 0
  for (int idx = t; idx < 128 * DDIM; idx += 256) {
    int row = idx >> 5, d = idx & 31;
    res[row >> 4][row & 15][d] =
        x[((size_t)blockIdx.x * 128 + row) * DDIM + d];
    dlt[row >> 4][row & 15][d] = 0.f;
  }
  __syncthreads();

  float lossAcc = 0.f;

  for (int i = 0; i < MSTAGE; ++i) {
    const float s  = scale[i];
    const float s2 = s * s;

    // ---- A fragment: this wave's 16 residual rows, bf16, contiguous k order
    // lane half h: K dims 16h..16h+15
    BF16x16 af;
    {
      const float* r0 = &res[w][lp][16 * h];
      const float* r1 = &res[w][lp][16 * h + 8];
#pragma unroll
      for (int e = 0; e < 8; ++e) af.s[e] = f2bf(r0[e]);
#pragma unroll
      for (int e = 0; e < 8; ++e) af.s[8 + e] = f2bf(r1[e]);
    }

    // ---- pass A: scores = 2*s*(r.c) - s^2*||c||^2 ; running max + argmax
    float mx[8]; int ax[8];
#pragma unroll
    for (int j = 0; j < 8; ++j) { mx[j] = -3.4e38f; ax[j] = 0; }

    for (int k0 = 0; k0 < KCB; k0 += 16) {
      // B fragment: column n=lp -> codebook row k0+lp, K dims 16h..16h+15
      BF16x16 bfm;
      const uint4* src = (const uint4*)(cbB + ((k0 + lp) * DDIM + 16 * h));
      bfm.q[0] = src[0]; bfm.q[1] = src[1];
      v8f cc = {};
      cc = __builtin_amdgcn_wmma_f32_16x16x32_bf16(false, af.v, false, bfm.v,
                                                   (short)0, cc, false, false);
      float c2 = cn2[k0 + lp];
#pragma unroll
      for (int j = 0; j < 8; ++j) {
        float sc = 2.f * s * cc[j] - s2 * c2;
        bool take = sc > mx[j];
        mx[j] = take ? sc : mx[j];
        ax[j] = take ? (k0 + lp) : ax[j];
      }
    }
    // reduce max/argmax across the 16 column lanes of each half (branchless)
#pragma unroll
    for (int off = 8; off >= 1; off >>= 1) {
#pragma unroll
      for (int j = 0; j < 8; ++j) {
        float om = __shfl_xor(mx[j], off);
        int   oa = __shfl_xor(ax[j], off);
        int gt = (om > mx[j]);
        int eq = (om == mx[j]);
        int lt = (oa < ax[j]);
        int take = gt | (eq & lt);          // no short-circuit -> cndmask only
        mx[j] = take ? om : mx[j];
        ax[j] = take ? oa : ax[j];
      }
    }
    if (lp == 0) {
#pragma unroll
      for (int j = 0; j < 8; ++j) codesS[w][i][8 * h + j] = ax[j];
    }
    __builtin_amdgcn_wave_barrier();

    // ---- pass B: p = exp(score - max); soft += P @ C via WMMA; l = sum p
    float lsum[8];
#pragma unroll
    for (int j = 0; j < 8; ++j) lsum[j] = 0.f;
    v8f cs0 = {}, cs1 = {};

    for (int k0c = 0; k0c < KCB; k0c += 32) {
#pragma unroll
      for (int tt = 0; tt < 2; ++tt) {
        int k0 = k0c + 16 * tt;
        BF16x16 bfm;
        const uint4* src = (const uint4*)(cbB + ((k0 + lp) * DDIM + 16 * h));
        bfm.q[0] = src[0]; bfm.q[1] = src[1];
        v8f cc = {};
        cc = __builtin_amdgcn_wmma_f32_16x16x32_bf16(false, af.v, false, bfm.v,
                                                     (short)0, cc, false, false);
        float c2 = cn2[k0 + lp];
#pragma unroll
        for (int j = 0; j < 8; ++j) {
          float sc = 2.f * s * cc[j] - s2 * c2;
          float p  = __expf(sc - mx[j]);             // mx matches row j+8h
          lsum[j] += p;
          pS[w][j + 8 * h][lp + 16 * tt] = p;        // transpose via LDS
        }
      }
      __builtin_amdgcn_wave_barrier();

      // B fragment of P^T: column n=lp (row index), K = 16h..16h+15
      BF16x16 bp;
      {
        const float* pp = &pS[w][lp][16 * h];
#pragma unroll
        for (int e = 0; e < 16; ++e) bp.s[e] = f2bf(pp[e]);
      }
      // A fragments: transposed codebook (rows = dims), contiguous k (16h..16h+15)
      BF16x16 a0, a1;
      {
        const unsigned short* t0 = cbT + lp * KCB + k0c;
        const unsigned short* t1 = cbT + (16 + lp) * KCB + k0c;
        a0.q[0] = *(const uint4*)(t0 + 16 * h);
        a0.q[1] = *(const uint4*)(t0 + 16 * h + 8);
        a1.q[0] = *(const uint4*)(t1 + 16 * h);
        a1.q[1] = *(const uint4*)(t1 + 16 * h + 8);
      }
      cs0 = __builtin_amdgcn_wmma_f32_16x16x32_bf16(false, a0.v, false, bp.v,
                                                    (short)0, cs0, false, false);
      cs1 = __builtin_amdgcn_wmma_f32_16x16x32_bf16(false, a1.v, false, bp.v,
                                                    (short)0, cs1, false, false);
      __builtin_amdgcn_wave_barrier();
    }

    // row sums of p
#pragma unroll
    for (int off = 8; off >= 1; off >>= 1) {
#pragma unroll
      for (int j = 0; j < 8; ++j) lsum[j] += __shfl_xor(lsum[j], off);
    }
    if (lp == 0) {
#pragma unroll
      for (int j = 0; j < 8; ++j) lS[w][8 * h + j] = lsum[j];
    }
    __builtin_amdgcn_wave_barrier();

    // soft = s * acc / l ; park in pS[row][d]  (cs frag: row=lp, d=16f+j+8h)
    {
      float linv = s / lS[w][lp];
#pragma unroll
      for (int j = 0; j < 8; ++j) {
        pS[w][lp][8 * h + j]      = cs0[j] * linv;
        pS[w][lp][16 + 8 * h + j] = cs1[j] * linv;
      }
    }
    __builtin_amdgcn_wave_barrier();

    // ---- residual / delta update + distortion accumulation
    // lane -> (row = lp, dims 16h..16h+15)
    {
      int code = codesS[w][i][lp];
      const unsigned short* ch = cbB + code * DDIM;
#pragma unroll
      for (int dd = 0; dd < 16; ++dd) {
        int d = 16 * h + dd;
        float hardv = s * bf2f(ch[d]);
        float r  = res[w][lp][d] - hardv;            // x - qhard_sum
        float sf = pS[w][lp][d];
        float dl = dlt[w][lp][d] + hardv - sf;       // qhard_sum - qsoft_sum
        res[w][lp][d] = r;
        dlt[w][lp][d] = dl;
        lossAcc += 0.1f * (r + dl) * (r + dl) + r * r;
        if (i == MSTAGE - 1) lossAcc += 0.1f * dl * dl;  // joint center
      }
    }
    __builtin_amdgcn_wave_barrier();
  }

  // ---- outputs: hard codes [N, M] as float, then scalar loss
  if (lane < 16) {
    size_t grow = (size_t)blockIdx.x * 128 + (size_t)w * 16 + lane;
#pragma unroll
    for (int i = 0; i < MSTAGE; ++i)
      out[grow * MSTAGE + i] = (float)codesS[w][i][lane];
  }
  float acc = lossAcc;
#pragma unroll
  for (int off = 16; off >= 1; off >>= 1) acc += __shfl_xor(acc, off);
  if (lane == 0)
    atomicAdd(out + (size_t)NROWS * MSTAGE,
              acc * (1.0f / ((float)NROWS * (float)DDIM)));
}

__global__ void drq_zero(float* p) {
  if (threadIdx.x == 0) p[0] = 0.f;
}

extern "C" void kernel_launch(void* const* d_in, const int* in_sizes, int n_in,
                              void* d_out, int out_size, void* d_ws, size_t ws_size,
                              hipStream_t stream) {
  (void)in_sizes; (void)n_in; (void)out_size; (void)d_ws; (void)ws_size;
  const float* x  = (const float*)d_in[0];
  const float* cb = (const float*)d_in[1];
  const float* sc = (const float*)d_in[2];
  float* out = (float*)d_out;

  drq_zero<<<1, 32, 0, stream>>>(out + (size_t)NROWS * MSTAGE);
  drq_kernel<<<NROWS / 128, 256, 0, stream>>>(x, cb, sc, out);
}
